// EdgeMLPPredictor_42597485641877
// MI455X (gfx1250) — matrix operand order repaired and checked by reading, not compiled, not run on hardware
//
#include <hip/hip_runtime.h>

// EdgeMLP predictor for gfx1250 (MI455X).
// GEMM1 (128x128x128 per 128-edge tile) runs on V_WMMA_F32_16X16X32_F16
// (f16 inputs, f32 accumulate): 16384 FLOP/instr vs 2048 for the f32 WMMA,
// which moves the kernel from matrix-pipe-bound to L2-gather-bound
// (h table = 25.6 MB, resident in the 192 MB L2).
// GEMM2 (N=2) + biases stay fp32 on VALU with a cross-lane shfl reduction.
//
// Fragment LDS layout (per CDNA5 ISA 7.12.2, 16-bit A 16x32 / B 32x16):
// each lane holds 16 f16 as two contiguous 8-f16 chunks. A fragment is
// stored as two 128-dword planes; lane L's chunk is at plane_base + L*16B,
// so a fragment fetch is two conflict-free ds_load_b128.
//   A (M=16,K=32): lane = half*16 + m;
//     plane0 chunk = A[m][8*half + 0..7],  plane1 chunk = A[m][16 + 8*half + 0..7]
//   B (K=32,N=16): lane = half*16 + n;
//     plane0 chunk = B[8*half + 0..7][n],  plane1 chunk = B[16 + 8*half + 0..7][n]
// All staging writes are ds_store_b128 (one fragment slot per store).

typedef _Float16 v8h  __attribute__((ext_vector_type(8)));
typedef _Float16 v16h __attribute__((ext_vector_type(16)));
typedef float    v8f  __attribute__((ext_vector_type(8)));

#define EMB    64
#define TILE_E 128   // edges per block
#define FRAG_H 512   // f16 elements per 16x32 / 32x16 fragment
#define PLANE  256   // f16 elements per fragment plane (128 dwords)

__global__ __launch_bounds__(256)
void edge_mlp_wmma_f16(const float* __restrict__ h,
                       const int*   __restrict__ edge_index,   // [2, E] int32
                       const float* __restrict__ W1,           // [128, 128]
                       const float* __restrict__ b1,           // [128]
                       const float* __restrict__ W2,           // [128, 2]
                       const float* __restrict__ b2,           // [2]
                       float*       __restrict__ out,          // [E, 2]
                       int E)
{
    extern __shared__ _Float16 smem_h[];
    _Float16* Xs  = smem_h;           // 8 mt * 4 kt * 512 = 16384 f16 (32 KB)
    _Float16* W1s = smem_h + 16384;   // 4 kt * 8 nt * 512 = 16384 f16 (32 KB)

    const int tid  = threadIdx.x;   // 0..255
    const int lane = tid & 31;
    const int wave = tid >> 5;      // 0..7
    const int l    = lane & 15;
    const int half = lane >> 4;

    // ---- stage W1 (f32 -> f16) into B-fragment layout ----
    // Thread owns a 16-row x 4-col patch: 16 coalesced float4 loads,
    // register transpose, 8 ds_store_b128 into fragment slots.
    {
        const int ng  = tid & 31;       // column group: n0 = ng*4
        const int kg  = tid >> 5;       // row group:    kb0 = kg*16
        const int n0  = ng * 4;
        const int kb0 = kg * 16;
        const int kt  = kg >> 1;        // k-tile 0..3
        const int bh  = kg & 1;         // plane owning these K rows (0-15 / 16-31)
        float4 vals[16];
        #pragma unroll
        for (int r = 0; r < 16; ++r)
            vals[r] = *(const float4*)(W1 + (kb0 + r) * 128 + n0);
        #pragma unroll
        for (int jj = 0; jj < 4; ++jj) {
            const int n = n0 + jj;
            _Float16* fb = W1s + ((kt * 8) + (n >> 4)) * FRAG_H
                               + bh * PLANE + (n & 15) * 8;
            #pragma unroll
            for (int p = 0; p < 2; ++p) {
                v8h cv;
                #pragma unroll
                for (int i = 0; i < 8; ++i) {
                    const float4 v = vals[p * 8 + i];
                    const float x = (jj == 0) ? v.x : (jj == 1) ? v.y
                                  : (jj == 2) ? v.z : v.w;
                    cv[i] = (_Float16)x;
                }
                *(v8h*)(fb + p * 16 * 8) = cv;
            }
        }
    }

    // ---- gather endpoint embeddings (f32 -> f16) into A-fragment layout ----
    const long long e0 = (long long)blockIdx.x * TILE_E;
    {
        const int eloc = tid >> 1;          // local edge 0..127
        const int side = tid & 1;           // 0 = src (k 0..63), 1 = dst (k 64..127)
        const long long ge  = e0 + eloc;
        const bool valid    = ge < (long long)E;
        const long long gec = valid ? ge : (long long)(E - 1);
        const int node = edge_index[(long long)side * E + gec];
        const float* hrow = h + (long long)node * EMB;
        const int mt_ = eloc >> 4, ml = eloc & 15;
        #pragma unroll
        for (int c = 0; c < 8; ++c) {       // {ktl, g, ah} chunk enumeration
            const int ktl = c >> 2;         // k-tile within this endpoint
            const int g   = (c >> 1) & 1;   // plane (K group 0-15 / 16-31)
            const int ah  = c & 1;          // lane half
            const int kb  = ktl * 32 + g * 16 + ah * 8;   // base K within endpoint
            float4 va, vb;
            if (valid) {
                va = *(const float4*)(hrow + kb);
                vb = *(const float4*)(hrow + kb + 4);
            } else {
                va = make_float4(0.f, 0.f, 0.f, 0.f);
                vb = va;
            }
            v8h cv;
            cv[0] = (_Float16)va.x; cv[1] = (_Float16)va.y;
            cv[2] = (_Float16)va.z; cv[3] = (_Float16)va.w;
            cv[4] = (_Float16)vb.x; cv[5] = (_Float16)vb.y;
            cv[6] = (_Float16)vb.z; cv[7] = (_Float16)vb.w;
            const int kt = side * 2 + ktl;  // global k-tile 0..3
            _Float16* dst = Xs + ((mt_ * 4) + kt) * FRAG_H
                               + g * PLANE
                               + (ah * 16 + ml) * 8;
            *(v8h*)dst = cv;
        }
    }

    // ---- per-lane epilogue constants (column n = nt*16 + l), full fp32 ----
    float  bias1[8];
    float2 w2v[8];
    #pragma unroll
    for (int j = 0; j < 8; ++j) {
        bias1[j] = b1[j * 16 + l];
        w2v[j]   = *(const float2*)(W2 + (j * 16 + l) * 2);
    }

    __syncthreads();

    // ---- GEMM1: hid[128x128] = X @ W1 via v_wmma_f32_16x16x32_f16 ----
    v8f acc[8];
    #pragma unroll
    for (int nt = 0; nt < 8; ++nt) acc[nt] = (v8f)0.f;

    const int mt = wave;                              // this wave's 16-edge strip
    const _Float16* aBase = Xs + (mt * 4) * FRAG_H + lane * 8;
    #pragma unroll
    for (int kt = 0; kt < 4; ++kt) {
        const v8h aLo = *(const v8h*)(aBase + kt * FRAG_H);
        const v8h aHi = *(const v8h*)(aBase + kt * FRAG_H + PLANE);
        const v16h a = __builtin_shufflevector(aLo, aHi,
            0, 1, 2, 3, 4, 5, 6, 7, 8, 9, 10, 11, 12, 13, 14, 15);
        const _Float16* bBase = W1s + (kt * 8) * FRAG_H + lane * 8;
        #pragma unroll
        for (int nt = 0; nt < 8; ++nt) {
            const v8h bLo = *(const v8h*)(bBase + nt * FRAG_H);
            const v8h bHi = *(const v8h*)(bBase + nt * FRAG_H + PLANE);
            const v16h b = __builtin_shufflevector(bLo, bHi,
                0, 1, 2, 3, 4, 5, 6, 7, 8, 9, 10, 11, 12, 13, 14, 15);
            acc[nt] = __builtin_amdgcn_wmma_f32_16x16x32_f16(
                false, a, false, b, (short)0, acc[nt], false, false);
        }
    }

    // ---- epilogue: bias + relu (fp32), then GEMM2 (N=2) on VALU ----
    // acc[nt][i] = hid[row = mt*16 + half*8 + i][col = nt*16 + l]
    float p0[8], p1[8];
    #pragma unroll
    for (int i = 0; i < 8; ++i) { p0[i] = 0.f; p1[i] = 0.f; }
    #pragma unroll
    for (int nt = 0; nt < 8; ++nt) {
        #pragma unroll
        for (int i = 0; i < 8; ++i) {
            float hv = acc[nt][i] + bias1[nt];
            hv = fmaxf(hv, 0.f);
            p0[i] = __builtin_fmaf(hv, w2v[nt].x, p0[i]);
            p1[i] = __builtin_fmaf(hv, w2v[nt].y, p1[i]);
        }
    }
    // reduce over the 16 lanes of each half (columns of hid)
    #pragma unroll
    for (int i = 0; i < 8; ++i) {
        #pragma unroll
        for (int m = 1; m <= 8; m <<= 1) {
            p0[i] += __shfl_xor(p0[i], m, 32);
            p1[i] += __shfl_xor(p1[i], m, 32);
        }
    }
    if (l == 0) {
        const float bb0 = b2[0], bb1 = b2[1];
        #pragma unroll
        for (int i = 0; i < 8; ++i) {
            const long long row = e0 + mt * 16 + half * 8 + i;
            if (row < (long long)E) {
                float2 o; o.x = p0[i] + bb0; o.y = p1[i] + bb1;
                *(float2*)(out + row * 2) = o;
            }
        }
    }
}

extern "C" void kernel_launch(void* const* d_in, const int* in_sizes, int n_in,
                              void* d_out, int out_size, void* d_ws, size_t ws_size,
                              hipStream_t stream) {
    const float* h  = (const float*)d_in[0];
    const int*   ei = (const int*)  d_in[1];   // [2, E] int32
    const float* W1 = (const float*)d_in[2];
    const float* b1 = (const float*)d_in[3];
    const float* W2 = (const float*)d_in[4];
    const float* b2 = (const float*)d_in[5];
    float* out = (float*)d_out;

    const int E = in_sizes[1] / 2;
    const int blocks = (E + TILE_E - 1) / TILE_E;
    const size_t shmem = 2u * 16384u * sizeof(_Float16);   // 64 KB dynamic LDS

    edge_mlp_wmma_f16<<<blocks, 256, shmem, stream>>>(h, ei, W1, b1, W2, b2, out, E);
}
